// Breadth_27315992002859
// MI455X (gfx1250) — matrix operand order repaired and checked by reading, not compiled, not run on hardware
//
#include <hip/hip_runtime.h>
#include <hip/hip_bf16.h>

typedef __attribute__((ext_vector_type(16))) __bf16 v16bf;
typedef __attribute__((ext_vector_type(8)))  float  v8f;

union ABFrag {
    v16bf v;
    uint4 q[2];
};

#define D_FEAT 96
#define EPS_NORM 1e-12f

// ---- order-preserving float<->uint mapping for atomic max on signed floats ----
__device__ __forceinline__ unsigned ford(float f) {
    unsigned u = __float_as_uint(f);
    return (u & 0x80000000u) ? ~u : (u | 0x80000000u);
}
__device__ __forceinline__ float funord(unsigned u) {
    unsigned b = (u & 0x80000000u) ? (u & 0x7FFFFFFFu) : ~u;
    return __uint_as_float(b);
}

__device__ __forceinline__ float pick8(const v8f& c, int i) {
    switch (i & 7) {
        case 0: return c[0];
        case 1: return c[1];
        case 2: return c[2];
        case 3: return c[3];
        case 4: return c[4];
        case 5: return c[5];
        case 6: return c[6];
        default: return c[7];
    }
}

// ---------------------------------------------------------------------------
// K1: row-wise L2 normalize, emit bf16 normalized features.
// One wave32 per row (3 features per lane).
// ---------------------------------------------------------------------------
__global__ void k_norm(const float* __restrict__ x, __bf16* __restrict__ xnbf, int N) {
    const int lane = threadIdx.x & 31;
    const int row  = blockIdx.x * 8 + (threadIdx.x >> 5);
    if (row >= N) return;
    const float* xr = x + (long)row * D_FEAT;
    float f0 = xr[lane];
    float f1 = xr[lane + 32];
    float f2 = xr[lane + 64];
    float s = f0 * f0 + f1 * f1 + f2 * f2;
    #pragma unroll
    for (int o = 16; o > 0; o >>= 1) s += __shfl_xor(s, o, 32);
    const float inv = 1.0f / fmaxf(sqrtf(s), EPS_NORM);
    __bf16* r = xnbf + (long)row * D_FEAT;
    r[lane]      = (__bf16)(f0 * inv);
    r[lane + 32] = (__bf16)(f1 * inv);
    r[lane + 64] = (__bf16)(f2 * inv);
}

// ---------------------------------------------------------------------------
// K2: attention logits via WMMA Gram-matrix diagonal.
// One wave handles 16 edges: A = xn[dst rows] (16x96), B = xn[src rows]^T
// (96x16), 3 x v_wmma_f32_16x16x32_bf16, diagonal = per-edge dot products.
//
// A frag (16x32 bf16): lane m = lane&15; kbase = lane<16 ? 0 : 8;
//   elems 0..7  = K kbase..kbase+7      (16B contiguous)
//   elems 8..15 = K kbase+16..kbase+23  (16B contiguous)
// B frag (32x16 bf16): lane col = lane&15; kbase = lane<16 ? 0 : 8;
//   elems 0..7 = K kbase..kbase+7, 8..15 = K kbase+16..kbase+23 (as A)  [B[k][n] = xn_src[n][k]]
// C/D (16x16 f32): vgpr r, lanes 0-15 -> (M=r,   N=lane),
//                          lanes 16-31 -> (M=r+8, N=lane-16).
//   diag m<8  owned by lane m,    elem m
//   diag m>=8 owned by lane m+16, elem m-8
// ---------------------------------------------------------------------------
__global__ void k_alpha(const __bf16* __restrict__ xnbf, const int* __restrict__ eidx,
                        const float* __restrict__ beta, float* __restrict__ alpha,
                        int E, int M) {
    const int  lane = threadIdx.x & 31;
    const long g    = (long)blockIdx.x * 8 + (threadIdx.x >> 5);
    const long g16  = g * 16;
    if (g16 >= M) return;  // wave-uniform exit: EXEC stays all-ones for WMMA

    const int role = lane & 15;
    long e = g16 + role;
    if (e >= M) e = g16;  // clamp padded slots to a valid edge (stores are guarded)

    // A-matrix row source: destination node of edge e; B-matrix column: source node
    const int dstA = (e < E) ? eidx[E + e] : (int)(e - E);
    const int srcB = (e < E) ? eidx[e]     : (int)(e - E);
    const int kbA = (lane < 16) ? 0 : 8;
    const int kbB = (lane < 16) ? 0 : 8;

    const __bf16* rowA = xnbf + (long)dstA * D_FEAT;
    const __bf16* rowB = xnbf + (long)srcB * D_FEAT;

    v8f c = {};
    #pragma unroll
    for (int kt = 0; kt < D_FEAT; kt += 32) {
        ABFrag a, b;
        const uint4* pa = (const uint4*)(rowA + kt + kbA);  // 16B aligned
        a.q[0] = pa[0];   // K = kbA + 0..7
        a.q[1] = pa[2];   // K = kbA + 16..23  (+16 elems = +2 uint4)
        const uint4* pb = (const uint4*)(rowB + kt + kbB);  // 16B aligned
        b.q[0] = pb[0];
        b.q[1] = pb[2];
        c = __builtin_amdgcn_wmma_f32_16x16x32_bf16(
                /*neg_a=*/false, a.v, /*neg_b=*/false, b.v,
                /*c_mod=*/(short)0, c, /*reuse_a=*/false, /*reuse_b=*/false);
    }

    const float b0 = beta[0];
    if (lane < 8) {                       // diag m = lane, elem = lane
        const long eo = g16 + lane;
        if (eo < M) alpha[eo] = b0 * pick8(c, lane);
    } else if (lane >= 24) {              // diag m = lane-16, elem = lane-24
        const long eo = g16 + (lane - 16);
        if (eo < M) alpha[eo] = b0 * pick8(c, lane - 24);
    }
}

// ---------------------------------------------------------------------------
// K3: segment max over destination nodes (orderable-uint atomic max).
// ---------------------------------------------------------------------------
__global__ void k_segmax(const int* __restrict__ eidx, const float* __restrict__ alpha,
                         unsigned* __restrict__ smax, int E, int M) {
    const long e = (long)blockIdx.x * 256 + threadIdx.x;
    if (e >= M) return;
    const int dst = (e < E) ? eidx[E + e] : (int)(e - E);
    atomicMax(smax + dst, ford(alpha[e]));
}

// ---------------------------------------------------------------------------
// K4: e = exp(alpha - segmax[dst]); denom[dst] += e; out[dst] += e * x[src]
// (normalization deferred to K5). One wave per edge, 3 features per lane.
// ---------------------------------------------------------------------------
__global__ void k_agg(const float* __restrict__ x, const int* __restrict__ eidx,
                      const float* __restrict__ alpha, const unsigned* __restrict__ smax,
                      float* __restrict__ denom, float* __restrict__ out,
                      int E, int M) {
    const int  lane = threadIdx.x & 31;
    const long e    = (long)blockIdx.x * 8 + (threadIdx.x >> 5);
    if (e >= M) return;
    const int src = (e < E) ? eidx[e]     : (int)(e - E);
    const int dst = (e < E) ? eidx[E + e] : (int)(e - E);
    const float m  = funord(smax[dst]);
    const float ee = __expf(alpha[e] - m);
    if (lane == 0) atomicAdd(denom + dst, ee);
    const float* xs = x   + (long)src * D_FEAT;
    float*       od = out + (long)dst * D_FEAT;
    #pragma unroll
    for (int d = lane; d < D_FEAT; d += 32) atomicAdd(od + d, ee * xs[d]);
}

// ---------------------------------------------------------------------------
// K5: out = tanh(out / denom[row])
// ---------------------------------------------------------------------------
__global__ void k_finish(float* __restrict__ out, const float* __restrict__ denom, long total) {
    const long i = (long)blockIdx.x * 256 + threadIdx.x;
    if (i >= total) return;
    out[i] = tanhf(out[i] / denom[i / D_FEAT]);
}

static inline size_t align256(size_t v) { return (v + 255) & ~(size_t)255; }

extern "C" void kernel_launch(void* const* d_in, const int* in_sizes, int n_in,
                              void* d_out, int out_size, void* d_ws, size_t ws_size,
                              hipStream_t stream) {
    const float* x    = (const float*)d_in[0];
    const float* beta = (const float*)d_in[1];
    const int*   eidx = (const int*)d_in[2];   // [2][E]: row 0 = src(j), row 1 = dst(i)

    const int N = in_sizes[0] / D_FEAT;
    const int E = in_sizes[2] / 2;
    const int M = E + N;                       // edges + self-loops

    // workspace layout
    char* w = (char*)d_ws;
    size_t off = 0;
    __bf16* xnbf = (__bf16*)(w + off);  off += align256((size_t)N * D_FEAT * sizeof(__bf16));
    float*  alpha = (float*)(w + off);  off += align256((size_t)M * sizeof(float));
    unsigned* smax = (unsigned*)(w + off); off += align256((size_t)N * sizeof(unsigned));
    float*  denom = (float*)(w + off);  off += align256((size_t)N * sizeof(float));
    (void)ws_size; (void)n_in;

    // zero accumulators (capture-safe async memsets)
    hipMemsetAsync(d_out, 0, (size_t)out_size * sizeof(float), stream);
    hipMemsetAsync(smax, 0, (size_t)N * sizeof(unsigned), stream);  // 0 < ford(any finite alpha)
    hipMemsetAsync(denom, 0, (size_t)N * sizeof(float), stream);

    // K1: normalize -> bf16   (8 waves/block, 1 row/wave)
    k_norm<<<(N + 7) / 8, 256, 0, stream>>>(x, xnbf, N);

    // K2: WMMA attention logits (8 waves/block, 16 edges/wave)
    const long groups = ((long)M + 15) / 16;
    k_alpha<<<(unsigned)((groups + 7) / 8), 256, 0, stream>>>(xnbf, eidx, beta, alpha, E, M);

    // K3: segment max
    k_segmax<<<(unsigned)(((long)M + 255) / 256), 256, 0, stream>>>(eidx, alpha, smax, E, M);

    // K4: exp + denom + unnormalized weighted scatter (1 wave/edge)
    k_agg<<<(unsigned)(((long)M + 7) / 8), 256, 0, stream>>>(x, eidx, alpha, smax, denom, d_out ? (float*)d_out : nullptr, E, M);

    // K5: normalize + tanh
    const long total = (long)N * D_FEAT;
    k_finish<<<(unsigned)((total + 255) / 256), 256, 0, stream>>>((float*)d_out, denom, total);
}
